// GNN2_36773509988941
// MI455X (gfx1250) — matrix operand order repaired and checked by reading, not compiled, not run on hardware
//
#include <hip/hip_runtime.h>

typedef __bf16 bf16;
typedef __attribute__((ext_vector_type(16))) __bf16 v16bf;
typedef __attribute__((ext_vector_type(8)))  __bf16 bf16x8;
typedef __attribute__((ext_vector_type(4)))  __bf16 bf16x4;
typedef __attribute__((ext_vector_type(8)))  float  v8f;
typedef __attribute__((ext_vector_type(4)))  unsigned int u32x4;
typedef __attribute__((ext_vector_type(4)))  int i32x4;

struct U32x8 { u32x4 lo; u32x4 hi; };

__device__ __forceinline__ float lrelu(float x) { return x > 0.f ? x : 0.01f * x; }

enum { EPI_PARTIAL = 0, EPI_LRELU = 1, EPI_LRELU_RES = 2, EPI_BF16T = 3 };

// ---- CDNA5 async global->LDS copy support (guarded; falls back to sync) ----
#if defined(__AMDGCN__) && __has_builtin(__builtin_amdgcn_global_load_async_to_lds_b128)
#define HAS_ASYNC_LDS 1
#else
#define HAS_ASYNC_LDS 0
#endif

// builtin signature: (int4 addrspace(1)* src, int4 addrspace(3)* dst, imm offset, imm cpol)
#define GPTR(p) ((__attribute__((address_space(1))) i32x4*)(p))
#define LPTR(p) ((__attribute__((address_space(3))) i32x4*)(p))

template<int N>
__device__ __forceinline__ void wait_async()
{
#if __has_builtin(__builtin_amdgcn_s_wait_asynccnt)
    __builtin_amdgcn_s_wait_asynccnt((unsigned short)N);
#else
    asm volatile("s_wait_asynccnt %0" :: "i"(N) : "memory");
#endif
}

// ---------------------------------------------------------------------------
// WMMA GEMM: C[M,N] = epi( A[M,K] @ Bt[N,K]^T + bias )
//   - A bf16 (AF32=false): async double-buffered LDS staging on CDNA5
//   - A fp32 (AF32=true, adjacency fallback): sync staging + in-register cvt
//   - Bt bf16 row-major [N][K]
//   - block tile 64 x BN, waves own 32x32 (2x2 v_wmma_f32_16x16x32_bf16)
//   - split-K over gridDim.z (EPI_PARTIAL -> per-chunk partials, deterministic)
// ---------------------------------------------------------------------------
template<int BN, bool AF32, int EPI>
__global__ __launch_bounds__(BN * 2)
void gemm_ker(const void* __restrict__ Aptr, const bf16* __restrict__ Bt,
              const float* __restrict__ bias, const float* __restrict__ res,
              void* __restrict__ Cptr, int M, int N, int K,
              int ldA, int ldC, int kChunk)
{
    constexpr int NT = BN * 2;
    constexpr int NBUF = (!AF32 && HAS_ASYNC_LDS) ? 2 : 1;
    // async ops issued per thread per tile
    constexpr int G = 256 / NT + (BN * 4) / NT;

    __shared__ __align__(16) bf16 sA[NBUF][64 * 32];
    __shared__ __align__(16) bf16 sB[NBUF][BN * 32];

    const int tid  = threadIdx.x;
    const int lane = tid & 31;
    const int wave = tid >> 5;
    const int wm   = wave & 1;        // 2 wave-rows over BM=64
    const int wn   = wave >> 1;       // BN/32 wave-cols
    const int l15  = lane & 15;
    const int half = lane >> 4;
    const int bm0  = blockIdx.x * 64;
    const int bn0  = blockIdx.y * BN;
    const int kb   = blockIdx.z * kChunk;
    const int keT  = kb + kChunk;
    const int ke   = keT < K ? keT : K;

    v8f acc[2][2] = {};

    auto compute = [&](int buf) {
        const u32x4* sA4 = (const u32x4*)sA[buf];
        const u32x4* sB4 = (const u32x4*)sB[buf];
        int ra0 = (wm * 32 + l15) * 4 + half;      // uint4 units, row = 4 uint4
        int ra1 = ra0 + 16 * 4;
        int rb0 = (wn * 32 + l15) * 4 + half;
        int rb1 = rb0 + 16 * 4;
        U32x8 ta0{sA4[ra0], sA4[ra0 + 2]};
        U32x8 ta1{sA4[ra1], sA4[ra1 + 2]};
        U32x8 tb0{sB4[rb0], sB4[rb0 + 2]};
        U32x8 tb1{sB4[rb1], sB4[rb1 + 2]};
        v16bf a0 = __builtin_bit_cast(v16bf, ta0);
        v16bf a1 = __builtin_bit_cast(v16bf, ta1);
        v16bf b0 = __builtin_bit_cast(v16bf, tb0);
        v16bf b1 = __builtin_bit_cast(v16bf, tb1);
        acc[0][0] = __builtin_amdgcn_wmma_f32_16x16x32_bf16(false, a0, false, b0, (short)0, acc[0][0], false, false);
        acc[0][1] = __builtin_amdgcn_wmma_f32_16x16x32_bf16(false, a0, false, b1, (short)0, acc[0][1], false, false);
        acc[1][0] = __builtin_amdgcn_wmma_f32_16x16x32_bf16(false, a1, false, b0, (short)0, acc[1][0], false, false);
        acc[1][1] = __builtin_amdgcn_wmma_f32_16x16x32_bf16(false, a1, false, b1, (short)0, acc[1][1], false, false);
    };

#if HAS_ASYNC_LDS
    if constexpr (!AF32) {
        const bf16* A = (const bf16*)Aptr;
        auto stage = [&](int kt, int buf) {
            #pragma unroll
            for (int s = tid; s < 256; s += NT) {
                int r = s >> 2, cs = (s & 3) * 8;
                __builtin_amdgcn_global_load_async_to_lds_b128(
                    GPTR(A + (size_t)(bm0 + r) * ldA + kt + cs),
                    LPTR(&sA[buf][r * 32 + cs]), 0, 0);
            }
            #pragma unroll
            for (int s = tid; s < BN * 4; s += NT) {
                int n = s >> 2, cs = (s & 3) * 8;
                __builtin_amdgcn_global_load_async_to_lds_b128(
                    GPTR(Bt + (size_t)(bn0 + n) * K + kt + cs),
                    LPTR(&sB[buf][n * 32 + cs]), 0, 0);
            }
        };
        stage(kb, 0);
        int buf = 0;
        for (int kt = kb; kt < ke; kt += 32) {
            if (kt + 32 < ke) {
                stage(kt + 32, buf ^ 1);   // prefetch next tile into other buffer
                wait_async<G>();           // current tile's copies done (in-order)
            } else {
                wait_async<0>();
            }
            __syncthreads();               // all waves' copies visible
            compute(buf);
            __syncthreads();               // done reading before buffer reuse
            buf ^= 1;
        }
    } else
#endif
    {
        for (int kt = kb; kt < ke; kt += 32) {
            __syncthreads();
            if constexpr (AF32) {
                const float* A = (const float*)Aptr;
                for (int s = tid; s < 256; s += NT) {
                    int r = s >> 2, cs = (s & 3) * 8;
                    const float* src = A + (size_t)(bm0 + r) * ldA + kt + cs;
                    float4 f0 = *(const float4*)src;
                    float4 f1 = *(const float4*)(src + 4);
                    bf16x8 v;
                    v[0] = (bf16)f0.x; v[1] = (bf16)f0.y; v[2] = (bf16)f0.z; v[3] = (bf16)f0.w;
                    v[4] = (bf16)f1.x; v[5] = (bf16)f1.y; v[6] = (bf16)f1.z; v[7] = (bf16)f1.w;
                    *(bf16x8*)&sA[0][r * 32 + cs] = v;
                }
            } else {
                const bf16* A = (const bf16*)Aptr;
                for (int s = tid; s < 256; s += NT) {
                    int r = s >> 2, cs = (s & 3) * 8;
                    u32x4 v = *(const u32x4*)(A + (size_t)(bm0 + r) * ldA + kt + cs);
                    *(u32x4*)&sA[0][r * 32 + cs] = v;
                }
            }
            for (int s = tid; s < BN * 4; s += NT) {
                int n = s >> 2, cs = (s & 3) * 8;
                u32x4 v = *(const u32x4*)(Bt + (size_t)(bn0 + n) * K + kt + cs);
                *(u32x4*)&sB[0][n * 32 + cs] = v;
            }
            __syncthreads();
            compute(0);
        }
    }

    // ---- epilogue (C layout: col = lane&15, row = vgpr + 8*(lane>=16)) ----
    #pragma unroll
    for (int i = 0; i < 2; ++i) {
        #pragma unroll
        for (int j = 0; j < 2; ++j) {
            int gmb = bm0 + wm * 32 + i * 16 + half * 8;
            int gc  = bn0 + wn * 32 + j * 16 + l15;
            if constexpr (EPI == EPI_PARTIAL) {
                float* out = (float*)Cptr + (size_t)blockIdx.z * M * ldC;
                #pragma unroll
                for (int e = 0; e < 8; ++e)
                    out[(size_t)(gmb + e) * ldC + gc] = acc[i][j][e];
            } else if constexpr (EPI == EPI_LRELU) {
                float bv = bias[gc];
                float* out = (float*)Cptr;
                #pragma unroll
                for (int e = 0; e < 8; ++e)
                    out[(size_t)(gmb + e) * ldC + gc] = lrelu(acc[i][j][e] + bv);
            } else if constexpr (EPI == EPI_LRELU_RES) {
                float bv = bias[gc];
                float* out = (float*)Cptr;
                #pragma unroll
                for (int e = 0; e < 8; ++e) {
                    size_t o = (size_t)(gmb + e) * ldC + gc;
                    out[o] = res[o] + lrelu(acc[i][j][e] + bv);
                }
            } else { // EPI_BF16T: bf16 transposed store (Mp^T), ldC = M
                float bv = bias[gc];
                bf16* out = (bf16*)Cptr;
                #pragma unroll
                for (int e = 0; e < 8; ++e)
                    out[(size_t)gc * ldC + (gmb + e)] = (bf16)(acc[i][j][e] + bv);
            }
        }
    }
}

// ---------------------------------------------------------------------------
// BatchNorm statistics: per-column mean and rsqrt(var+eps), biased variance.
// ---------------------------------------------------------------------------
__global__ void bn_stats(const float* __restrict__ X, int Nrows, int C,
                         float* __restrict__ mean, float* __restrict__ rstd)
{
    __shared__ float ssum[256];
    __shared__ float ssq[256];
    int c    = blockIdx.x * 64 + (threadIdx.x & 63);
    int part = threadIdx.x >> 6;
    float s = 0.f, q = 0.f;
    for (int r = part; r < Nrows; r += 4) {
        float v = X[(size_t)r * C + c];
        s += v; q += v * v;
    }
    ssum[threadIdx.x] = s;
    ssq[threadIdx.x]  = q;
    __syncthreads();
    if (part == 0) {
        #pragma unroll
        for (int p = 1; p < 4; ++p) { s += ssum[threadIdx.x + 64 * p]; q += ssq[threadIdx.x + 64 * p]; }
        float m = s / (float)Nrows;
        float v = q / (float)Nrows - m * m;
        mean[c] = m;
        rstd[c] = rsqrtf(v + 1e-5f);
    }
}

// normalize + affine + convert to bf16 (feeds WMMA A operand)
__global__ void norm_cvt(const float* __restrict__ X, const float* __restrict__ mean,
                         const float* __restrict__ rstd, const float* __restrict__ g,
                         const float* __restrict__ be, bf16* __restrict__ Y,
                         long long total, int cmask)
{
    long long i = ((long long)blockIdx.x * blockDim.x + threadIdx.x) * 4;
    if (i >= total) return;
    float4 x = *(const float4*)(X + i);
    int c = (int)(i & cmask);
    bf16x4 y;
    y[0] = (bf16)((x.x - mean[c + 0]) * rstd[c + 0] * g[c + 0] + be[c + 0]);
    y[1] = (bf16)((x.y - mean[c + 1]) * rstd[c + 1] * g[c + 1] + be[c + 1]);
    y[2] = (bf16)((x.z - mean[c + 2]) * rstd[c + 2] * g[c + 2] + be[c + 2]);
    y[3] = (bf16)((x.w - mean[c + 3]) * rstd[c + 3] * g[c + 3] + be[c + 3]);
    *(bf16x4*)(Y + i) = y;
}

// plain fp32 -> bf16 conversion (features, adjacency pre-pass)
__global__ void cvt_bf16(const float* __restrict__ X, bf16* __restrict__ Y, long long total)
{
    long long i = ((long long)blockIdx.x * blockDim.x + threadIdx.x) * 4;
    if (i >= total) return;
    float4 x = *(const float4*)(X + i);
    bf16x4 y;
    y[0] = (bf16)x.x; y[1] = (bf16)x.y; y[2] = (bf16)x.z; y[3] = (bf16)x.w;
    *(bf16x4*)(Y + i) = y;
}

// feat[nt,128] -> cat[:,0:128] (cat row stride 256)
__global__ void copy2cat(const float* __restrict__ feat, float* __restrict__ cat, long long total)
{
    long long i = ((long long)blockIdx.x * blockDim.x + threadIdx.x) * 4;
    if (i >= total) return;
    int r = (int)(i >> 7), c = (int)(i & 127);
    *(float4*)(cat + (size_t)r * 256 + c) = *(const float4*)(feat + i);
}

// sum 8 split-K partials [8][nt][128] -> cat[:,128:256]
__global__ void reduce_part(const float* __restrict__ part, float* __restrict__ cat,
                            long long total)
{
    long long i = ((long long)blockIdx.x * blockDim.x + threadIdx.x) * 4;
    if (i >= total) return;
    float4 s = *(const float4*)(part + i);
    #pragma unroll
    for (int z = 1; z < 8; ++z) {
        float4 v = *(const float4*)(part + (size_t)z * total + i);
        s.x += v.x; s.y += v.y; s.z += v.z; s.w += v.w;
    }
    int r = (int)(i >> 7), c = (int)(i & 127);
    *(float4*)(cat + (size_t)r * 256 + 128 + c) = s;
}

// W[K,O] fp32 -> Wt[O,K] bf16 (transposed, contiguous-K for the B operand)
__global__ void wcvt(const float* __restrict__ W, bf16* __restrict__ Wt, int K, int O)
{
    int idx = blockIdx.x * 256 + threadIdx.x;
    if (idx >= K * O) return;
    int o = idx / K, k = idx - o * K;
    Wt[idx] = (bf16)W[(size_t)k * O + o];
}

// ---------------------------------------------------------------------------
// Host side
// ---------------------------------------------------------------------------
namespace {
struct Lin { const float* W; const float* b; int K, O; bf16* Wt; };
struct Mlp { const float *g1, *be1, *g2, *be2; Lin l1, l2; int i, h, o; };
struct Msg { Lin lin; Mlp mlp; };
}

extern "C" void kernel_launch(void* const* d_in, const int* in_sizes, int n_in,
                              void* d_out, int out_size, void* d_ws, size_t ws_size,
                              hipStream_t stream)
{
    (void)in_sizes; (void)n_in; (void)out_size;

    // ---- workspace bump allocator ----
    char* ws = (char*)d_ws;
    size_t off = 0;
    auto alloc = [&](size_t bytes) -> void* {
        void* p = ws + off;
        off = (off + bytes + 255) & ~(size_t)255;
        return p;
    };
    float* Hm   = (float*)alloc((size_t)8192 * 128 * 4);
    float* Ha   = (float*)alloc((size_t)4096 * 128 * 4);
    float* Hd   = (float*)alloc((size_t)4096 * 128 * 4);
    float* cat  = (float*)alloc((size_t)8192 * 256 * 4);
    float* hid  = (float*)alloc((size_t)8192 * 512 * 4);
    bf16*  bst  = (bf16*)alloc((size_t)8192 * 512 * 2);
    bf16*  MpT  = (bf16*)alloc((size_t)128 * 8192 * 2);
    float* part = (float*)alloc((size_t)8 * 8192 * 128 * 4);
    float* stats = (float*)alloc((size_t)1024 * 4);
    bf16*  wts  = (bf16*)alloc((size_t)2801664 * 2);

    const float* adjF[4] = { (const float*)d_in[3], (const float*)d_in[4],
                             (const float*)d_in[5], (const float*)d_in[6] };
    // pre-converted bf16 adjacency (if workspace allows): halves hot-loop bytes
    const size_t adjElems = (size_t)4096 * 8192;
    bf16* adjB[4] = { nullptr, nullptr, nullptr, nullptr };
    bool pre = (off + 4 * adjElems * 2) <= ws_size;
    if (pre) {
        for (int a = 0; a < 4; ++a) {
            adjB[a] = (bf16*)alloc(adjElems * 2);
            long long tot = (long long)adjElems;
            cvt_bf16<<<(int)((tot / 4 + 255) / 256), 256, 0, stream>>>(adjF[a], adjB[a], tot);
        }
    }

    // ---- parse params in setup_inputs() insertion order ----
    int cur = 7;
    bf16* wp = wts;
    auto nextp = [&]() { return (const float*)d_in[cur++]; };
    auto mkLin = [&](int K, int O) {
        Lin l; l.W = nextp(); l.b = nextp(); l.K = K; l.O = O; l.Wt = wp;
        wp += (size_t)K * O;
        int n = K * O;
        wcvt<<<(n + 255) / 256, 256, 0, stream>>>(l.W, l.Wt, K, O);
        return l;
    };
    auto mkMlp = [&](int i, int o) {
        Mlp m; m.i = i; m.h = i * 2; m.o = o;
        m.g1 = nextp(); m.be1 = nextp();
        m.l1 = mkLin(i, m.h);
        m.g2 = nextp(); m.be2 = nextp();
        m.l2 = mkLin(m.h, o);
        return m;
    };
    auto mkMsg = [&]() { Msg s; s.lin = mkLin(128, 128); s.mlp = mkMlp(256, 128); return s; };

    Mlp inM  = mkMlp(128, 128);
    Mlp inA  = mkMlp(128, 128);
    Mlp inD  = mkMlp(128, 128);
    Mlp outM = mkMlp(128, 64);
    Msg blocks[3][4];
    for (int l = 0; l < 3; ++l)
        for (int k = 0; k < 4; ++k)
            blocks[l][k] = mkMsg();

    // ---- building blocks ----
    auto runMlp = [&](const Mlp& m, const float* X, int N, float* out, const float* res) {
        bn_stats<<<m.i / 64, 256, 0, stream>>>(X, N, m.i, stats, stats + m.i);
        long long t1 = (long long)N * m.i;
        norm_cvt<<<(int)((t1 / 4 + 255) / 256), 256, 0, stream>>>(X, stats, stats + m.i, m.g1, m.be1, bst, t1, m.i - 1);
        gemm_ker<128, false, EPI_LRELU><<<dim3(N / 64, m.h / 128, 1), 256, 0, stream>>>(
            bst, m.l1.Wt, m.l1.b, nullptr, hid, N, m.h, m.i, m.i, m.h, m.i);
        bn_stats<<<m.h / 64, 256, 0, stream>>>(hid, N, m.h, stats, stats + m.h);
        long long t2 = (long long)N * m.h;
        norm_cvt<<<(int)((t2 / 4 + 255) / 256), 256, 0, stream>>>(hid, stats, stats + m.h, m.g2, m.be2, bst, t2, m.h - 1);
        if (m.o % 128 == 0) {
            if (res)
                gemm_ker<128, false, EPI_LRELU_RES><<<dim3(N / 64, m.o / 128, 1), 256, 0, stream>>>(
                    bst, m.l2.Wt, m.l2.b, res, out, N, m.o, m.h, m.h, m.o, m.h);
            else
                gemm_ker<128, false, EPI_LRELU><<<dim3(N / 64, m.o / 128, 1), 256, 0, stream>>>(
                    bst, m.l2.Wt, m.l2.b, nullptr, out, N, m.o, m.h, m.h, m.o, m.h);
        } else {
            gemm_ker<64, false, EPI_LRELU><<<dim3(N / 64, m.o / 64, 1), 128, 0, stream>>>(
                bst, m.l2.Wt, m.l2.b, nullptr, out, N, m.o, m.h, m.h, m.o, m.h);
        }
    };

    auto runMsg = [&](const Msg& g, const float* src, int ns, float* feat, int nt,
                      const float* adj_f32, const bf16* adj_b16) {
        long long ts = (long long)ns * 128;
        cvt_bf16<<<(int)((ts / 4 + 255) / 256), 256, 0, stream>>>(src, bst, ts);
        // Mp^T[128, ns] = (src @ Wlin + b)^T, stored bf16 (B operand of the agg GEMM)
        gemm_ker<128, false, EPI_BF16T><<<dim3(ns / 64, 1, 1), 256, 0, stream>>>(
            bst, g.lin.Wt, g.lin.b, nullptr, MpT, ns, 128, 128, 128, ns, 128);
        // M2N = adj @ Mp : split-K=8 deterministic partials
        if (adj_b16)
            gemm_ker<128, false, EPI_PARTIAL><<<dim3(nt / 64, 1, 8), 256, 0, stream>>>(
                adj_b16, MpT, nullptr, nullptr, part, nt, 128, ns, ns, 128, ns / 8);
        else
            gemm_ker<128, true, EPI_PARTIAL><<<dim3(nt / 64, 1, 8), 256, 0, stream>>>(
                adj_f32, MpT, nullptr, nullptr, part, nt, 128, ns, ns, 128, ns / 8);
        long long tf = (long long)nt * 128;
        copy2cat<<<(int)((tf / 4 + 255) / 256), 256, 0, stream>>>(feat, cat, tf);
        reduce_part<<<(int)((tf / 4 + 255) / 256), 256, 0, stream>>>(part, cat, tf);
        // feat = feat + mlp(cat)   (residual fused in final GEMM epilogue)
        runMlp(g.mlp, cat, nt, feat, feat);
    };

    // ---- full forward pass ----
    runMlp(inM, (const float*)d_in[0], 8192, Hm, nullptr);
    runMlp(inA, (const float*)d_in[1], 4096, Ha, nullptr);
    runMlp(inD, (const float*)d_in[2], 4096, Hd, nullptr);

    for (int l = 0; l < 3; ++l) {
        runMsg(blocks[l][0], Hm, 8192, Ha, 4096, adjF[1], adjB[1]);  // A <- m2a
        runMsg(blocks[l][1], Hm, 8192, Hd, 4096, adjF[0], adjB[0]);  // D <- m2d
        runMsg(blocks[l][2], Hd, 4096, Hm, 8192, adjF[2], adjB[2]);  // M <- d2m
        runMsg(blocks[l][3], Ha, 4096, Hm, 8192, adjF[3], adjB[3]);  // M <- a2m
    }

    runMlp(outM, Hm, 8192, (float*)d_out, nullptr);
}
